// TriRasterizer_48137993453605
// MI455X (gfx1250) — matrix operand order repaired and checked by reading, not compiled, not run on hardware
//
#include <hip/hip_runtime.h>
#include <cstdint>
#include <cstddef>

#define RH 1080
#define RW 1920

typedef __attribute__((ext_vector_type(16))) _Float16 v16h;
typedef __attribute__((ext_vector_type(8)))  float    v8f;
typedef int v4i __attribute__((vector_size(16)));
typedef __attribute__((address_space(1))) v4i GV4;
typedef __attribute__((address_space(3))) v4i LV4;

// ---- CDNA5 async global->LDS path (ASYNCcnt), with safe fallback ----
#if defined(__has_builtin)
#if __has_builtin(__builtin_amdgcn_global_load_async_to_lds_b128)
#define HAVE_ASYNC_LDS 1
#endif
#endif

#ifdef HAVE_ASYNC_LDS
#if __has_builtin(__builtin_amdgcn_s_wait_asynccnt)
#define ASYNC_WAIT0() __builtin_amdgcn_s_wait_asynccnt(0)
#else
#define ASYNC_WAIT0() asm volatile("s_wait_asynccnt 0" ::: "memory")
#endif
#else
#define ASYNC_WAIT0()
#endif

__device__ __forceinline__ void rastAsyncB128(const void* g, void* l) {
#ifdef HAVE_ASYNC_LDS
  __builtin_amdgcn_global_load_async_to_lds_b128((GV4*)(v4i*)(void*)g, (LV4*)(v4i*)l, 0, 0);
#else
  *(float4*)l = *(const float4*)g;
#endif
}

// ---------------- kernel 0: init output (=backdrop) and count plane ----------------
__global__ void rast_init(float* __restrict__ out, const float* __restrict__ bg,
                          float* __restrict__ cnt, int hwc, int hw) {
  const int stride = gridDim.x * blockDim.x;
  for (int i = blockIdx.x * blockDim.x + threadIdx.x; i < hwc; i += stride) out[i] = bg[i];
  for (int i = blockIdx.x * blockDim.x + threadIdx.x; i < hw; i += stride) cnt[i] = 0.0f;
}

// ---------------- kernel 1: per-triangle quantized area ----------------
__global__ __launch_bounds__(256) void rast_area(const float* __restrict__ tri,
                                                 int* __restrict__ area, int T) {
  const int t = blockIdx.x * 256 + threadIdx.x;
  if (t >= T) return;
  const float* p = tri + (size_t)t * 6;
  const float x0 = p[0], y0 = p[1], x1 = p[2], y1 = p[3], x2 = p[4], y2 = p[5];
  const float a = fabsf(0.5f * (x0 * (y1 - y2) + x1 * (y2 - y0) + x2 * (y0 - y1)));
  area[t] = (int)(a * 0.1f);  // truncation toward zero, matches astype(int32)
}

// ---------------- kernel 2: inclusive prefix sum via WMMA ----------------
// One block = 8 wave32 waves. Each wave scans 256 ints per chunk with ONE
// v_wmma_f32_16x16x32_f16:  D = L(16x16 lower-tri ones, f16) x X(16 cols of 16).
// Exact arithmetic: inputs are ints <= 80 (exact in f16 <= 2048); running sum
// ~3.5M < 2^24 (exact in f32). Cross-segment/chunk carry via LDS.
__global__ __launch_bounds__(256) void rast_scan(const int* __restrict__ area,
                                                 int* __restrict__ cum, int T) {
  __shared__ float segtot[128];
  __shared__ float segoff[128];
  __shared__ float carry;
  const int tid  = threadIdx.x;
  const int wave = tid >> 5;
  const int lane = tid & 31;
  if (tid == 0) carry = 0.0f;
  __syncthreads();

  // A = lower-triangular ones (K<16), zeros for K>=16.
  // 16-bit A layout (contiguous): lane m (<16) holds K=0..15 in halves 0..15;
  // lane m+16 holds K=16..31 (zeros).
  v16h A = {};
  {
    const int m = lane & 15;
    if (lane < 16) { for (int i = 0; i < 16; ++i) A[i] = (_Float16)((i     <= m) ? 1.0f : 0.0f); }
  }

  const int nChunks = (T + 2047) >> 11;        // 2048 elements per block-chunk
  for (int ch = 0; ch < nChunks; ++ch) {
    const int wbase = (ch << 11) + wave * 256; // this wave's 256 elements
    // B layout (32x16 f16): lane n<16 halves i hold B[i][n] (K=0..15);
    // lanes 16..31 hold K=16..31 -> zero padding rows.
    v16h B = {};
    if (lane < 16) {
      const int segbase = wbase + lane * 16;   // column n = segment n
      for (int i = 0; i < 16; ++i) {
        const int idx = segbase + i;
        B[i] = (_Float16)((idx < T) ? (float)area[idx] : 0.0f);
      }
    }
    v8f Cz = {};
    // EXEC is all-ones here (uniform control flow) as WMMA requires.
    v8f D = __builtin_amdgcn_wmma_f32_16x16x32_f16(false, A, false, B, (short)0, Cz,
                                                   false, false);
    // D layout: lane l<16: D[r][l]; lane l>=16: D[8+r][l-16]. Segment total = D[15][n].
    if (lane >= 16) segtot[wave * 16 + (lane - 16)] = D[7];
    __syncthreads();
    if (tid == 0) {                             // tiny serial combine of 128 totals
      float run = carry;
      for (int i = 0; i < 128; ++i) { segoff[i] = run; run += segtot[i]; }
      carry = run;
    }
    __syncthreads();
    const int seg   = lane & 15;
    const float off = segoff[wave * 16 + seg];
    const int mbase = (lane >= 16) ? 8 : 0;
    const int ebase = wbase + seg * 16 + mbase;
    for (int r = 0; r < 8; ++r) {
      const int idx = ebase + r;
      if (idx < T) cum[idx] = (int)(D[r] + off + 0.5f);
    }
    __syncthreads();                            // segtot reused next chunk
  }
}

// ---------------- kernel 3: point emission + scatter ----------------
// One thread per triangle; per-triangle slots [cum-a, cum) reproduce
// searchsorted(cum, pos, 'right') exactly. Triangle data staged to LDS via
// the async copy engine (15 KB per 256-triangle block).
__global__ __launch_bounds__(256) void rast_emit(
    const float* __restrict__ tri, const float* __restrict__ fea,
    const int* __restrict__ area, const int* __restrict__ cum,
    float* __restrict__ out, float* __restrict__ cnt, int T) {
  __shared__ float sTri[256 * 6];   // 6 KB
  __shared__ float sFea[256 * 9];   // 9 KB
  const int tid       = threadIdx.x;
  const int blockBase = blockIdx.x * 256;
  const int rem       = T - blockBase;
  const int nTri      = rem < 256 ? rem : 256;
  const int vB = nTri * 24;         // multiple of 16 (nTri is 256 or 96)
  const int fB = nTri * 36;         // multiple of 16
  const char* gv = (const char*)(tri + (size_t)blockBase * 6);
  const char* gf = (const char*)(fea + (size_t)blockBase * 9);
  for (int off = tid * 16; off < vB; off += 4096) rastAsyncB128(gv + off, ((char*)sTri) + off);
  for (int off = tid * 16; off < fB; off += 4096) rastAsyncB128(gf + off, ((char*)sFea) + off);
  ASYNC_WAIT0();
  __syncthreads();
  if (tid >= nTri) return;

  const int t = blockBase + tid;
  const float* tp = sTri + tid * 6;
  const float x0 = tp[0], y0 = tp[1], x1 = tp[2], y1 = tp[3], x2 = tp[4], y2 = tp[5];
  // geometric height -> int16; reciprocal computed in fp16 (reference quirk)
  const float dxg = x2 - (x0 + x1) * 0.5f;
  const float dyg = y2 - (y0 + y1) * 0.5f;
  const short hC   = (short)sqrtf(dxg * dxg + dyg * dyg);
  const float hCf  = (float)hC;
  const float recC = (float)((_Float16)1.0f / (_Float16)hCf);
  // int16-truncated vertices for the coordinate fill
  const float p0x = (float)(short)x0, p0y = (float)(short)y0;
  const float p1x = (float)(short)x1, p1y = (float)(short)y1;
  const float p2x = (float)(short)x2, p2y = (float)(short)y2;

  const float* fp = sFea + tid * 9;
  float f0[3], f1[3], f2[3];
  for (int c = 0; c < 3; ++c) { f0[c] = fp[c]; f1[c] = fp[3 + c]; f2[c] = fp[6 + c]; }
  // feature-space "height": _fill reuses channels 0/1 of the features
  const float gx = f2[0] - (f0[0] + f1[0]) * 0.5f;
  const float gy = f2[1] - (f0[1] + f1[1]) * 0.5f;
  const short hF   = (short)sqrtf(gx * gx + gy * gy);
  const float hFf  = (float)hF;
  const float recF = (float)((_Float16)1.0f / (_Float16)hFf);

  const int a    = area[t];
  const int cg   = cum[t];
  const int base = cg - a;
  const float cgf = (float)cg;
  for (int j = 0; j < a; ++j) {
    const int pos = base + j;
    const float normalized = 1.0f - (float)(cg - pos) / cgf;
    const float ex = 1.0f - sqrtf(1.0f - normalized);
    // coordinates
    const float ijc_ = fmodf(ex, recC);
    const float ijc  = ijc_ * hCf;
    const float kc   = ex - ijc_;
    const float omi  = 1.0f - ijc, omk = 1.0f - kc;
    const float cx = p2x * kc + (p0x * omi + p1x * ijc) * omk;
    const float cy = p2y * kc + (p0y * omi + p1y * ijc) * omk;
    int xi = (int)cx, yi = (int)cy;
    xi = xi < 0 ? 0 : (xi > RW - 1 ? RW - 1 : xi);
    yi = yi < 0 ? 0 : (yi > RH - 1 ? RH - 1 : yi);
    // features
    const float ijf_ = fmodf(ex, recF);
    const float ijf  = ijf_ * hFf;
    const float kf   = ex - ijf_;
    const float omfi = 1.0f - ijf, omfk = 1.0f - kf;
    const size_t pix = (size_t)yi * RW + xi;
    float* o = out + pix * 3;
    for (int c = 0; c < 3; ++c)                     // arbitrary-winner overwrite
      o[c] = f2[c] * kf + (f0[c] * omfi + f1[c] * ijf) * omfk;
    atomicAdd(&cnt[pix], 1.0f);                     // global_atomic_add_f32
  }
}

// ---------------- kernel 4: normalize (0/0 -> NaN like the reference) ----------------
__global__ __launch_bounds__(256) void rast_norm(float* __restrict__ out,
                                                 const float* __restrict__ cnt, int hw) {
  const int i = blockIdx.x * 256 + threadIdx.x;
  if (i >= hw) return;
  const float c = cnt[i];
  float* o = out + (size_t)i * 3;
  o[0] /= c; o[1] /= c; o[2] /= c;
}

extern "C" void kernel_launch(void* const* d_in, const int* in_sizes, int n_in,
                              void* d_out, int out_size, void* d_ws, size_t ws_size,
                              hipStream_t stream) {
  (void)n_in; (void)out_size; (void)ws_size;
  const float* tri = (const float*)d_in[0];   // [T,3,2] f32
  const float* fea = (const float*)d_in[1];   // [T,3,3] f32
  const float* bg  = (const float*)d_in[2];   // [H,W,3] f32
  float* out = (float*)d_out;                 // [H,W,3] f32
  const int T = in_sizes[0] / 6;

  int*   d_area = (int*)d_ws;                 // T ints
  int*   d_cum  = d_area + T;                 // T ints
  float* d_cnt  = (float*)(d_cum + T);        // H*W floats  (~10.7 MB total)

  const int hw = RH * RW, hwc = hw * 3;
  rast_init<<<2048, 256, 0, stream>>>(out, bg, d_cnt, hwc, hw);
  rast_area<<<(T + 255) / 256, 256, 0, stream>>>(tri, d_area, T);
  rast_scan<<<1, 256, 0, stream>>>(d_area, d_cum, T);
  rast_emit<<<(T + 255) / 256, 256, 0, stream>>>(tri, fea, d_area, d_cum, out, d_cnt, T);
  rast_norm<<<(hw + 255) / 256, 256, 0, stream>>>(out, d_cnt, hw);
}
